// BitLinear_55147380080918
// MI455X (gfx1250) — 8-bit matrix operand order repaired and checked by reading, compiled, not run on hardware
//
#include <hip/hip_runtime.h>
#include <stdint.h>

#define QBF 128.0f
#define EPSF 1e-5f

typedef int v8i __attribute__((ext_vector_type(8)));

static constexpr int M_TOK = 8192;
static constexpr int KDIM  = 4096;
static constexpr int NDIM  = 4096;

__global__ void bl_wabs_partial(const float* __restrict__ w, float* __restrict__ partial) {
    __shared__ float red[256];
    const int row = blockIdx.x;
    const float4* w4 = (const float4*)(w + (size_t)row * KDIM);
    float s = 0.f;
#pragma unroll
    for (int i = 0; i < 4; ++i) {
        float4 v = w4[threadIdx.x + i * 256];
        s += fabsf(v.x) + fabsf(v.y) + fabsf(v.z) + fabsf(v.w);
    }
    red[threadIdx.x] = s;
    __syncthreads();
    for (int off = 128; off > 0; off >>= 1) {
        if ((int)threadIdx.x < off) red[threadIdx.x] += red[threadIdx.x + off];
        __syncthreads();
    }
    if (threadIdx.x == 0) partial[row] = red[0];
}

__global__ void bl_wabs_final(const float* __restrict__ partial, float* __restrict__ scale_w) {
    __shared__ float red[256];
    float s = 0.f;
#pragma unroll
    for (int i = 0; i < 16; ++i) s += partial[threadIdx.x + i * 256];
    red[threadIdx.x] = s;
    __syncthreads();
    for (int off = 128; off > 0; off >>= 1) {
        if ((int)threadIdx.x < off) red[threadIdx.x] += red[threadIdx.x + off];
        __syncthreads();
    }
    if (threadIdx.x == 0) scale_w[0] = red[0] * (1.0f / 16777216.0f) + EPSF;
}

__global__ void bl_quant_w(const float* __restrict__ w, const float* __restrict__ scale_w,
                           int* __restrict__ wq_packed) {
    const float inv = 1.0f / scale_w[0];
    const int gid = blockIdx.x * 256 + threadIdx.x;
    const float4* w4 = (const float4*)w;
#pragma unroll
    for (int j = 0; j < 4; ++j) {
        const int idx = gid + j * 1048576;
        float4 v = w4[idx];
        float q0 = fminf(fmaxf(rintf(v.x * inv), -1.f), 1.f);
        float q1 = fminf(fmaxf(rintf(v.y * inv), -1.f), 1.f);
        float q2 = fminf(fmaxf(rintf(v.z * inv), -1.f), 1.f);
        float q3 = fminf(fmaxf(rintf(v.w * inv), -1.f), 1.f);
        int packed = ((int)q0 & 255) | (((int)q1 & 255) << 8) |
                     (((int)q2 & 255) << 16) | (((int)q3 & 255) << 24);
        wq_packed[idx] = packed;
    }
}

__global__ void bl_quant_x(const float* __restrict__ x, const float* __restrict__ scale_w,
                           int* __restrict__ xq_packed, float* __restrict__ combined) {
    __shared__ float red[256];
    const int row = blockIdx.x;
    const float4* x4 = (const float4*)(x + (size_t)row * KDIM);
    float m = 0.f;
#pragma unroll
    for (int i = 0; i < 4; ++i) {
        float4 v = x4[threadIdx.x + i * 256];
        m = fmaxf(m, fmaxf(fmaxf(fabsf(v.x), fabsf(v.y)), fmaxf(fabsf(v.z), fabsf(v.w))));
    }
    red[threadIdx.x] = m;
    __syncthreads();
    for (int off = 128; off > 0; off >>= 1) {
        if ((int)threadIdx.x < off) red[threadIdx.x] = fmaxf(red[threadIdx.x], red[threadIdx.x + off]);
        __syncthreads();
    }
    if (threadIdx.x == 0) {
        float sx = red[0] + EPSF;
        red[0] = sx;
        combined[row] = scale_w[0] * sx * (1.0f / QBF);
    }
    __syncthreads();
    const float s = QBF / red[0];
    int* xq_row = xq_packed + (size_t)row * (KDIM / 4);
#pragma unroll
    for (int i = 0; i < 4; ++i) {
        const int idx = threadIdx.x + i * 256;
        float4 v = x4[idx];
        float q0 = fminf(fmaxf(rintf(v.x * s), -127.f), 127.f);
        float q1 = fminf(fmaxf(rintf(v.y * s), -127.f), 127.f);
        float q2 = fminf(fmaxf(rintf(v.z * s), -127.f), 127.f);
        float q3 = fminf(fmaxf(rintf(v.w * s), -127.f), 127.f);
        int packed = ((int)q0 & 255) | (((int)q1 & 255) << 8) |
                     (((int)q2 & 255) << 16) | (((int)q3 & 255) << 24);
        xq_row[idx] = packed;
    }
}

__global__ void __launch_bounds__(256, 2)
bl_gemm_iu8(const int8_t* __restrict__ xq, const int8_t* __restrict__ wq,
            const float* __restrict__ combined, float* __restrict__ out) {
    const int lane  = threadIdx.x & 31;
    const int wave  = threadIdx.x >> 5;
    const int wy    = wave >> 2;
    const int wx    = wave & 3;
    const int waveM = blockIdx.y * 128 + wy * 64;
    const int waveN = blockIdx.x * 128 + wx * 32;
    const int row   = lane & 15;
    const int khalf = lane >> 4;

    v8i acc[4][2];
#pragma unroll
    for (int mt = 0; mt < 4; ++mt)
#pragma unroll
        for (int nt = 0; nt < 2; ++nt)
            acc[mt][nt] = (v8i)(0);

    const int8_t* aBase[4];
#pragma unroll
    for (int mt = 0; mt < 4; ++mt)
        aBase[mt] = xq + (size_t)(waveM + mt * 16 + row) * KDIM + khalf * 16;
    const int8_t* bBase[2];
#pragma unroll
    for (int nt = 0; nt < 2; ++nt)
        bBase[nt] = wq + (size_t)(waveN + nt * 16 + row) * KDIM + khalf * 16;

    for (int k = 0; k < KDIM; k += 64) {
        v8i a[4], b[2];
#pragma unroll
        for (int mt = 0; mt < 4; ++mt) {
            const int8_t* p = aBase[mt] + k;
            int2 c0 = *(const int2*)(p);
            int2 c1 = *(const int2*)(p + 8);
            int2 c2 = *(const int2*)(p + 32);
            int2 c3 = *(const int2*)(p + 40);
            v8i av = {c0.x, c0.y, c1.x, c1.y, c2.x, c2.y, c3.x, c3.y};
            a[mt] = av;
        }
#pragma unroll
        for (int nt = 0; nt < 2; ++nt) {
            const int8_t* p = bBase[nt] + k;
            int4 lo = *(const int4*)(p);
            int4 hi = *(const int4*)(p + 32);
            v8i bv = {lo.x, lo.y, lo.z, lo.w, hi.x, hi.y, hi.z, hi.w};
            b[nt] = bv;
        }
#pragma unroll
        for (int mt = 0; mt < 4; ++mt)
#pragma unroll
            for (int nt = 0; nt < 2; ++nt)
                acc[mt][nt] = __builtin_amdgcn_wmma_i32_16x16x64_iu8(
                     true, a[mt],  true, b[nt],
                    acc[mt][nt],  false,  false);
    }

    const int ncol0 = waveN + (lane & 15);
    const int mhi   = (lane >> 4) * 8;
#pragma unroll
    for (int mt = 0; mt < 4; ++mt) {
        const int mrow0 = waveM + mt * 16 + mhi;
        float cmb[8];
#pragma unroll
        for (int r = 0; r < 8; ++r) cmb[r] = combined[mrow0 + r];
#pragma unroll
        for (int nt = 0; nt < 2; ++nt) {
            const size_t base = (size_t)mrow0 * NDIM + ncol0 + nt * 16;
#pragma unroll
            for (int r = 0; r < 8; ++r)
                out[base + (size_t)r * NDIM] = (float)acc[mt][nt][r] * cmb[r];
        }
    }
}

extern "C" void kernel_launch(void* const* d_in, const int* in_sizes, int n_in,
                              void* d_out, int out_size, void* d_ws, size_t ws_size,
                              hipStream_t stream) {
    (void)in_sizes; (void)n_in; (void)out_size; (void)ws_size;

    const float* x = (const float*)d_in[0];
    const float* w = (const float*)d_in[1];
    float* out = (float*)d_out;

    char* ws = (char*)d_ws;
    int8_t* wq       = (int8_t*)ws;
    int8_t* xq       = (int8_t*)(ws + (size_t)16 * 1024 * 1024);
    float*  combined = (float*) (ws + (size_t)48 * 1024 * 1024);
    float*  partial  = (float*) (ws + (size_t)48 * 1024 * 1024 + 64 * 1024);
    float*  scale_w  = (float*) (ws + (size_t)48 * 1024 * 1024 + 128 * 1024);

    bl_wabs_partial<<<4096, 256, 0, stream>>>(w, partial);
    bl_wabs_final  <<<1,    256, 0, stream>>>(partial, scale_w);
    bl_quant_w     <<<4096, 256, 0, stream>>>(w, scale_w, (int*)wq);
    bl_quant_x     <<<M_TOK,256, 0, stream>>>(x, scale_w, (int*)xq, combined);

    dim3 grid(NDIM / 128, M_TOK / 128);
    bl_gemm_iu8<<<grid, dim3(256), 0, stream>>>(xq, wq, combined, out);
}
